// GAT_40149354283202
// MI455X (gfx1250) — matrix operand order repaired and checked by reading, not compiled, not run on hardware
//
#include <hip/hip_runtime.h>

#define NM_ 100000
#define NT_ 100000
#define ND_ 50000
#define EDGES_ 1000000
#define DIN_ 128

typedef __attribute__((ext_vector_type(16))) __bf16 v16bf;
typedef __attribute__((ext_vector_type(8)))  float  v8f;

union FragBF { uint4 q[2]; v16bf v; };

__device__ __forceinline__ unsigned short f2bf(float f) {
  unsigned u = __float_as_uint(f);
  u += 0x7FFFu + ((u >> 16) & 1u);   // round-to-nearest-even
  return (unsigned short)(u >> 16);
}

// order-preserving float -> u32 map (for atomic max on signed floats)
__device__ __forceinline__ unsigned mapf(float f) {
  unsigned u = __float_as_uint(f);
  return (u & 0x80000000u) ? ~u : (u | 0x80000000u);
}
__device__ __forceinline__ float unmapf(unsigned u) {
  return (u & 0x80000000u) ? __uint_as_float(u ^ 0x80000000u)
                           : __uint_as_float(~u);
}

__global__ void k_fill_f32(float* __restrict__ p, float v, int n) {
  int i = blockIdx.x * blockDim.x + threadIdx.x;
  if (i < n) p[i] = v;
}
__global__ void k_fill_u32(unsigned* __restrict__ p, unsigned v, int n) {
  int i = blockIdx.x * blockDim.x + threadIdx.x;
  if (i < n) p[i] = v;
}
__global__ void k_cast_bf16(const float* __restrict__ x, unsigned short* __restrict__ y, int n) {
  int i = blockIdx.x * blockDim.x + threadIdx.x;
  if (i < n) y[i] = f2bf(x[i]);
}
// out[n, j] = b1[j] + b2[j]  (hetero-sum of the two biases targeting this dst type)
__global__ void k_bias_init(const float* __restrict__ b1, const float* __restrict__ b2,
                            float* __restrict__ out, int n, int hf) {
  int i = blockIdx.x * blockDim.x + threadIdx.x;
  if (i < n * hf) { int j = i % hf; out[i] = b1[j] + b2[j]; }
}
// Wt[c, k] = bf16(W[k, c]) : transpose+convert so B-fragments are row-contiguous
__global__ void k_wt(const float* __restrict__ W, unsigned short* __restrict__ Wt, int HF) {
  int i = blockIdx.x * blockDim.x + threadIdx.x;
  if (i >= DIN_ * HF) return;
  int k = i / HF, c = i - k * HF;
  Wt[(size_t)c * DIN_ + k] = f2bf(W[i]);
}
// ul[k,h] = sum_f W[k, h*F+f] * al[h,f]   (ditto ur with ar) -> el = x @ ul exactly
__global__ void k_make_u(const float* __restrict__ W, const float* __restrict__ al,
                         const float* __restrict__ ar, float* __restrict__ ul,
                         float* __restrict__ ur, int H, int F) {
  int i = blockIdx.x * blockDim.x + threadIdx.x;
  if (i >= DIN_ * H) return;
  int k = i / H, h = i - k * H;
  int HF = H * F;
  float sl = 0.f, sr = 0.f;
  for (int f = 0; f < F; ++f) {
    float w = W[(size_t)k * HF + h * F + f];
    sl += w * al[h * F + f];
    sr += w * ar[h * F + f];
  }
  ul[k * H + h] = sl;
  ur[k * H + h] = sr;
}
// out[n,h] = sum_k X[n,k] * U[k,h]   (skinny GEMV, U cached in LDS)
__global__ void k_scores(const float* __restrict__ X, const float* __restrict__ U,
                         float* __restrict__ out, int N, int H) {
  __shared__ float su[DIN_ * 4];
  for (int i = threadIdx.x; i < DIN_ * H; i += blockDim.x) su[i] = U[i];
  __syncthreads();
  int n = blockIdx.x * blockDim.x + threadIdx.x;
  if (n >= N) return;
  const float* x = X + (size_t)n * DIN_;
  float acc[4] = {0.f, 0.f, 0.f, 0.f};
  for (int k = 0; k < DIN_; ++k) {
    float xv = x[k];
#pragma unroll 4
    for (int h = 0; h < H; ++h) acc[h] += xv * su[k * H + h];
  }
  for (int h = 0; h < H; ++h) out[(size_t)n * H + h] = acc[h];
}

// C[N,HF] = A[N,128](bf16) x Wt^T (Wt is [HF,128] bf16), f32 accumulate via WMMA.
// One wave per 16x16 tile; block = 4 waves covering 64 columns; K-loop 4 x 16x16x32.
__global__ void k_gemm_wmma(const unsigned short* __restrict__ A,
                            const unsigned short* __restrict__ Wt,
                            float* __restrict__ C, int N, int HF) {
  const int lane = threadIdx.x & 31;
  const int wave = threadIdx.x >> 5;
  const int half = lane >> 4;     // K half-range selector (ISA 16-bit A/B layouts)
  const int ln   = lane & 15;     // M row (A) / N col (B,C)
  const int r0 = blockIdx.y * 16;
  const int c0 = blockIdx.x * 64 + wave * 16;
  if (c0 >= HF) return;           // wave-uniform
  v8f acc = {0.f, 0.f, 0.f, 0.f, 0.f, 0.f, 0.f, 0.f};
  const unsigned short* arow = A  + (size_t)(r0 + ln) * DIN_;
  const unsigned short* brow = Wt + (size_t)(c0 + ln) * DIN_;
#pragma unroll
  for (int k0 = 0; k0 < DIN_; k0 += 32) {
    FragBF fa, fb;
    // A 16x32: dwords 0-3 = K[half*8 .. +7], dwords 4-7 = K[16+half*8 .. +7]
    fa.q[0] = *(const uint4*)(arow + k0 + half * 8);
    fa.q[1] = *(const uint4*)(arow + k0 + 16 + half * 8);
    // B 32x16: dwords 0-3 = K[half*8 .. +7], dwords 4-7 = K[16+half*8 .. +7] of column c0+ln
    fb.q[0] = *(const uint4*)(brow + k0 + half * 8);
    fb.q[1] = *(const uint4*)(brow + k0 + 16 + half * 8);
    acc = __builtin_amdgcn_wmma_f32_16x16x32_bf16(
        false, fa.v, false, fb.v, (short)0, acc, false, false);
  }
  // C 16x16 f32: VGPR v -> row r0 + half*8 + v, col c0 + ln
  float* crow = C + (size_t)(r0 + half * 8) * HF + c0 + ln;
#pragma unroll
  for (int v = 0; v < 8; ++v) crow[(size_t)v * HF] = acc[v];
}

__device__ __forceinline__ float edge_logit(const float* __restrict__ el,
                                            const float* __restrict__ er,
                                            int s, int d, int H, int h) {
  float v = el[(size_t)s * H + h] + er[(size_t)d * H + h];
  return v > 0.f ? v : 0.2f * v;  // leaky_relu, slope 0.2
}

__global__ void k_edge_max(const int* __restrict__ src, const int* __restrict__ dst,
                           const float* __restrict__ el, const float* __restrict__ er,
                           unsigned* __restrict__ mu, int E, int H) {
  int t = blockIdx.x * blockDim.x + threadIdx.x;
  if (t >= E * H) return;
  int e = t / H, h = t - e * H;
  int s = src[e], d = dst[e];
  float v = edge_logit(el, er, s, d, H, h);
  atomicMax(&mu[(size_t)d * H + h], mapf(v));
}

__global__ void k_edge_sum(const int* __restrict__ src, const int* __restrict__ dst,
                           const float* __restrict__ el, const float* __restrict__ er,
                           const unsigned* __restrict__ mu, float* __restrict__ den,
                           int E, int H) {
  int t = blockIdx.x * blockDim.x + threadIdx.x;
  if (t >= E * H) return;
  int e = t / H, h = t - e * H;
  int s = src[e], d = dst[e];
  float v = edge_logit(el, er, s, d, H, h);
  float m = unmapf(mu[(size_t)d * H + h]);
  atomicAdd(&den[(size_t)d * H + h], __expf(v - m));
}

__global__ void k_edge_aggr(const int* __restrict__ src, const int* __restrict__ dst,
                            const float* __restrict__ el, const float* __restrict__ er,
                            const unsigned* __restrict__ mu, const float* __restrict__ den,
                            const float* __restrict__ fs, float* __restrict__ out,
                            int E, int H, int F) {
  int t = blockIdx.x * blockDim.x + threadIdx.x;
  if (t >= E * H) return;
  int e = t / H, h = t - e * H;
  int s = src[e], d = dst[e];
  float v = edge_logit(el, er, s, d, H, h);
  float m = unmapf(mu[(size_t)d * H + h]);
  float a = __expf(v - m) / fmaxf(den[(size_t)d * H + h], 1e-12f);
  const float* fr = fs + (size_t)s * (H * F) + h * F;
  float* orow = out + (size_t)d * (H * F) + h * F;
  for (int f = 0; f < F; ++f) atomicAdd(&orow[f], fr[f] * a);
}

__global__ void k_elu(const float* __restrict__ x, float* __restrict__ y, int n) {
  int i = blockIdx.x * blockDim.x + threadIdx.x;
  if (i < n) { float v = x[i]; y[i] = v > 0.f ? v : expm1f(v); }
}

extern "C" void kernel_launch(void* const* d_in, const int* in_sizes, int n_in,
                              void* d_out, int out_size, void* d_ws, size_t ws_size,
                              hipStream_t stream) {
  (void)in_sizes; (void)n_in; (void)out_size; (void)ws_size;
  const float* xin[3] = {(const float*)d_in[0], (const float*)d_in[1], (const float*)d_in[2]};
  const float* Wl[3]  = {(const float*)d_in[3], (const float*)d_in[7],  (const float*)d_in[11]};
  const float* All[3] = {(const float*)d_in[4], (const float*)d_in[8],  (const float*)d_in[12]};
  const float* Arl[3] = {(const float*)d_in[5], (const float*)d_in[9],  (const float*)d_in[13]};
  const float* Bl[3]  = {(const float*)d_in[6], (const float*)d_in[10], (const float*)d_in[14]};
  const int* srcAll = (const int*)d_in[15];
  const int* dstAll = (const int*)d_in[16];

  const int    NN[3]   = {NM_, NT_, ND_};
  const size_t noff[3] = {0, NM_, (size_t)NM_ + NT_};
  const int rs[6] = {0, 1, 0, 2, 1, 2};
  const int rd[6] = {1, 0, 2, 0, 2, 1};
  const int bp[3][2] = {{1, 3}, {0, 5}, {2, 4}};  // relations feeding each dst type

  char* wp = (char*)d_ws;
  auto alloc = [&](size_t bytes) -> void* {
    void* r = (void*)wp; wp += (bytes + 255) & ~(size_t)255; return r;
  };
  const size_t TOTN = (size_t)NM_ + NT_ + ND_;
  float*          hA  = (float*)alloc(TOTN * DIN_ * sizeof(float));
  float*          hB  = (float*)alloc(TOTN * DIN_ * sizeof(float));
  unsigned short* hbf = (unsigned short*)alloc(TOTN * DIN_ * sizeof(unsigned short));
  float*          fsb = (float*)alloc((size_t)NM_ * DIN_ * sizeof(float));
  unsigned short* wt  = (unsigned short*)alloc((size_t)DIN_ * DIN_ * sizeof(unsigned short));
  float*          ul  = (float*)alloc(DIN_ * 4 * sizeof(float));
  float*          ur  = (float*)alloc(DIN_ * 4 * sizeof(float));
  float*          el  = (float*)alloc((size_t)NM_ * 4 * sizeof(float));
  float*          er  = (float*)alloc((size_t)NM_ * 4 * sizeof(float));
  unsigned*       mu  = (unsigned*)alloc((size_t)NM_ * 4 * sizeof(unsigned));
  float*          den = (float*)alloc((size_t)NM_ * 4 * sizeof(float));

  for (int l = 0; l < 3; ++l) {
    const int H = (l == 2) ? 1 : 4;
    const int F = (l == 2) ? 64 : 32;
    const int HF = H * F;
    const float* in_t[3];
    float* out_t[3];
    for (int t = 0; t < 3; ++t) {
      in_t[t]  = (l == 0) ? xin[t] : hA + noff[t] * DIN_;
      out_t[t] = (l == 2) ? ((float*)d_out + noff[t] * 64) : (hB + noff[t] * DIN_);
    }
    for (int t = 0; t < 3; ++t) {  // bf16 copies of current inputs
      int n = NN[t] * DIN_;
      k_cast_bf16<<<(n + 255) / 256, 256, 0, stream>>>(in_t[t], hbf + noff[t] * DIN_, n);
    }
    for (int t = 0; t < 3; ++t) {  // out = sum of biases of incoming relations
      int n = NN[t] * HF;
      k_bias_init<<<(n + 255) / 256, 256, 0, stream>>>(
          Bl[l] + (size_t)bp[t][0] * HF, Bl[l] + (size_t)bp[t][1] * HF, out_t[t], NN[t], HF);
    }
    for (int r = 0; r < 6; ++r) {
      const int s = rs[r], d = rd[r];
      const float* W  = Wl[l]  + (size_t)r * DIN_ * HF;
      const float* al = All[l] + (size_t)r * HF;
      const float* ar = Arl[l] + (size_t)r * HF;
      int nw = DIN_ * HF;
      k_wt<<<(nw + 255) / 256, 256, 0, stream>>>(W, wt, HF);
      k_make_u<<<(DIN_ * H + 127) / 128, 128, 0, stream>>>(W, al, ar, ul, ur, H, F);
      k_scores<<<(NN[s] + 255) / 256, 256, 0, stream>>>(in_t[s], ul, el, NN[s], H);
      k_scores<<<(NN[d] + 255) / 256, 256, 0, stream>>>(in_t[d], ur, er, NN[d], H);
      dim3 gg((HF + 63) / 64, NN[s] / 16);  // node counts are multiples of 16
      k_gemm_wmma<<<gg, 128, 0, stream>>>(hbf + noff[s] * DIN_, wt, fsb, NN[s], HF);
      int ndh = NN[d] * H;
      k_fill_u32<<<(ndh + 255) / 256, 256, 0, stream>>>(mu, 0u, ndh);
      k_fill_f32<<<(ndh + 255) / 256, 256, 0, stream>>>(den, 0.f, ndh);
      const int* srcp = srcAll + (size_t)r * EDGES_;
      const int* dstp = dstAll + (size_t)r * EDGES_;
      int eh = EDGES_ * H;
      k_edge_max<<<(eh + 255) / 256, 256, 0, stream>>>(srcp, dstp, el, er, mu, EDGES_, H);
      k_edge_sum<<<(eh + 255) / 256, 256, 0, stream>>>(srcp, dstp, el, er, mu, den, EDGES_, H);
      k_edge_aggr<<<(eh + 255) / 256, 256, 0, stream>>>(srcp, dstp, el, er, mu, den, fsb,
                                                        out_t[d], EDGES_, H, F);
    }
    if (l < 2) {
      int n = (int)(TOTN * DIN_);
      k_elu<<<(n + 255) / 256, 256, 0, stream>>>(hB, hA, n);
    }
  }
}
